// MGULayer_32641751449643
// MI455X (gfx1250) — matrix operand order repaired and checked by reading, not compiled, not run on hardware
//
#include <hip/hip_runtime.h>
#include <hip/hip_bf16.h>

// ---------------------------------------------------------------------------
// MGU recurrence on MI455X (gfx1250), wave32 + v_wmma_f32_16x16x32_bf16.
//   Phase 1: convert/transpose weights to bf16 [NxK], zero h.
//   Phase 2: time-parallel projections xf/xc via WMMA (bf16 in, f32 acc).
//   Phase 3: 512 sequential steps, 2 kernels each. Each block stages the whole
//            h/f*h A-matrix (128KB) + its 64-col U_f/U_c strip (128KB) into
//            WGP LDS via GLOBAL_LOAD_ASYNC_TO_LDS_B128, then 8 waves run the
//            WMMA GEMM out of LDS. U_f/U_c bf16 stay L2-resident all 512 steps.
// ---------------------------------------------------------------------------

typedef __bf16 bf16_t;
typedef __attribute__((ext_vector_type(16))) __bf16 v16bf;
typedef __attribute__((ext_vector_type(8)))  __bf16 v8bf;
typedef __attribute__((ext_vector_type(8)))  float  v8f;
typedef __attribute__((ext_vector_type(4)))  float  v4f;
typedef __attribute__((ext_vector_type(4)))  int    v4i;

#define B_DIM 64
#define T_DIM 512
#define D_DIM 512
#define U_DIM 1024
#define MROWS (B_DIM * T_DIM)   // 32768 rows for the projection GEMM

#define LDS_A_BYTES (B_DIM * U_DIM * 2)   // 131072: full h (or f*h) tile
#define LDS_B_BYTES (64 * U_DIM * 2)      // 131072: 64-column strip of U^T
#define STEP_LDS_BYTES (LDS_A_BYTES + LDS_B_BYTES)  // 256KB of the 320KB WGP LDS

// ---- CDNA5 async global->LDS copy (ASYNCcnt), with safe fallback ----------
#if defined(__has_builtin)
#if __has_builtin(__builtin_amdgcn_global_load_async_to_lds_b128)
#define HAVE_ASYNC_LDS 1
#endif
#endif
#ifndef HAVE_ASYNC_LDS
#define HAVE_ASYNC_LDS 0
#endif

__device__ __forceinline__ void async_copy16(void* lds_dst, const void* gsrc) {
#if HAVE_ASYNC_LDS
  __builtin_amdgcn_global_load_async_to_lds_b128(
      (__attribute__((address_space(1))) v4i*)(uintptr_t)gsrc,
      (__attribute__((address_space(3))) v4i*)(uintptr_t)lds_dst,
      0, 0);
#else
  *(v8bf*)lds_dst = *(const v8bf*)gsrc;
#endif
}

__device__ __forceinline__ void wait_async0() {
#if HAVE_ASYNC_LDS
#if __has_builtin(__builtin_amdgcn_s_wait_asynccnt)
  __builtin_amdgcn_s_wait_asynccnt(0);
#else
  asm volatile("s_wait_asynccnt 0x0" ::: "memory");
#endif
#endif
}

__device__ __forceinline__ bf16_t f2bf(float f) {
  unsigned u = __builtin_bit_cast(unsigned, f);
  unsigned r = u + 0x7FFFu + ((u >> 16) & 1u);   // round-to-nearest-even
  unsigned short h = (unsigned short)(r >> 16);
  return __builtin_bit_cast(bf16_t, h);
}

// A fragment (16x32 bf16, MxK) from a row-major bf16 matrix (global or LDS).
// lane<16 -> row M=lane, K = kb..kb+15;
// lane>=16 -> row M=lane-16, K = kb+16..kb+31 (matches load_b_t).
__device__ __forceinline__ v16bf load_a_bf16(const bf16_t* A, int lda,
                                             int m0, int kb0, int lane) {
  const int r = lane & 15;
  const int hlf = lane >> 4;
  const bf16_t* p = A + (size_t)(m0 + r) * lda + kb0 + hlf * 16;
  v8bf lo = *(const v8bf*)(p);
  v8bf hi = *(const v8bf*)(p + 8);
  v16bf out;
#pragma unroll
  for (int i = 0; i < 8; ++i) { out[i] = lo[i]; out[i + 8] = hi[i]; }
  return out;
}

// A fragment with on-the-fly f32 -> bf16 conversion (projection GEMM).
__device__ __forceinline__ v16bf load_a_f32(const float* __restrict__ A,
                                            int lda, int m0, int kb0, int lane) {
  const int r = lane & 15;
  const int hlf = lane >> 4;
  const float* p = A + (size_t)(m0 + r) * lda + kb0 + hlf * 16;
  v4f a0 = *(const v4f*)(p);
  v4f a1 = *(const v4f*)(p + 4);
  v4f a2 = *(const v4f*)(p + 8);
  v4f a3 = *(const v4f*)(p + 12);
  v16bf out;
#pragma unroll
  for (int i = 0; i < 4; ++i) {
    out[i]      = f2bf(a0[i]);
    out[i + 4]  = f2bf(a1[i]);
    out[i + 8]  = f2bf(a2[i]);
    out[i + 12] = f2bf(a3[i]);
  }
  return out;
}

// B fragment (32x16 bf16, KxN) from a PRE-TRANSPOSED [N x K] matrix:
// lane<16 -> col N=lane, K=kb..kb+15 ; lane>=16 -> col N=lane-16, K=kb+16..kb+31.
// Contiguous 32-byte load per lane.
__device__ __forceinline__ v16bf load_b_t(const bf16_t* BT, int ldb,
                                          int kb0, int n0, int lane) {
  const int n = n0 + (lane & 15);
  const int kb = kb0 + ((lane >> 4) ? 16 : 0);
  return *(const v16bf*)(BT + (size_t)n * ldb + kb);
}

// ---------------------------------------------------------------------------
// Weight convert + transpose: W [K x N] f32 -> WT [N x K] bf16 (one-time).
__global__ void mgu_cvt_wt(const float* __restrict__ W, bf16_t* __restrict__ WT,
                           int K, int N) {
  size_t total = (size_t)K * N;
  size_t stride = (size_t)gridDim.x * blockDim.x;
  for (size_t o = (size_t)blockIdx.x * blockDim.x + threadIdx.x; o < total; o += stride) {
    size_t n = o / (size_t)K;
    size_t k = o % (size_t)K;
    WT[o] = f2bf(W[k * (size_t)N + n]);
  }
}

__global__ void mgu_init_h(float* __restrict__ h_f32, bf16_t* __restrict__ h_bf) {
  int i = blockIdx.x * blockDim.x + threadIdx.x;
  if (i < B_DIM * U_DIM) { h_f32[i] = 0.0f; h_bf[i] = f2bf(0.0f); }
}

// ---------------------------------------------------------------------------
// Projection GEMM: out[(b*T+t)][u] = X @ W + bias, for both f and c (grid.z).
// Each wave: one 16x64 output strip, K=512 in 16 WMMA steps per N-tile.
__global__ __launch_bounds__(256) void mgu_proj(
    const float* __restrict__ X,       // [32768 x 512]
    const bf16_t* __restrict__ WfT,    // [1024 x 512]
    const bf16_t* __restrict__ WcT,    // [1024 x 512]
    const float* __restrict__ bias_f,
    const float* __restrict__ bias_c,
    float* __restrict__ xf,            // [32768 x 1024]
    float* __restrict__ xc) {
  const int lane = threadIdx.x & 31;
  const int wave = threadIdx.x >> 5;
  const int m0 = (blockIdx.y * 8 + wave) * 16;   // 2048 M-tiles
  const int n0 = blockIdx.x * 64;                // 16 N-strips
  const bf16_t* WT   = blockIdx.z ? WcT : WfT;
  const float*  bias = blockIdx.z ? bias_c : bias_f;
  float*        out  = blockIdx.z ? xc : xf;

  v8f acc[4] = {};
  for (int kb0 = 0; kb0 < D_DIM; kb0 += 32) {
    v16bf a = load_a_f32(X, D_DIM, m0, kb0, lane);
#pragma unroll
    for (int j = 0; j < 4; ++j) {
      v16bf bfrag = load_b_t(WT, D_DIM, kb0, n0 + j * 16, lane);
      acc[j] = __builtin_amdgcn_wmma_f32_16x16x32_bf16(
          false, a, false, bfrag, (short)0, acc[j], false, false);
    }
  }
  const int hlf = lane >> 4;
  const int cidx = lane & 15;
#pragma unroll
  for (int j = 0; j < 4; ++j) {
    const int col = n0 + j * 16 + cidx;
    const float bv = bias[col];
#pragma unroll
    for (int v = 0; v < 8; ++v) {
      const int row = m0 + v + hlf * 8;        // C/D layout: lane>=16 -> M+8
      out[(size_t)row * U_DIM + col] = acc[j][v] + bv;
    }
  }
}

// ---------------------------------------------------------------------------
// Stage A (full [64 x 1024] bf16 tile) + B (64-col strip of U^T) into LDS.
// Both source regions are contiguous 128KB blocks -> pure b128 async streams.
__device__ __forceinline__ void stage_step_tiles(const bf16_t* __restrict__ gA,
                                                 const bf16_t* __restrict__ gB,
                                                 char* smem, int tid) {
#pragma unroll 4
  for (int i = 0; i < 32; ++i) {
    const int idx = i * 256 + tid;                    // b128 index
    async_copy16(smem + (size_t)idx * 16, gA + (size_t)idx * 8);
    async_copy16(smem + LDS_A_BYTES + (size_t)idx * 16, gB + (size_t)idx * 8);
  }
  wait_async0();
  __syncthreads();
}

// Step kernel 1: f = sigmoid(xf_t + h @ U_f); emit f (f32) and f*h (bf16).
// Grid: 16 blocks x 8 waves; block owns 64 output columns, all 64 batch rows.
__global__ __launch_bounds__(256) void mgu_step_f(
    int t,
    const float* __restrict__ xf,      // [B*T x U]
    const bf16_t* __restrict__ h_bf,   // [64 x 1024]
    const float* __restrict__ h_f32,   // [64 x 1024]
    const bf16_t* __restrict__ UfT,    // [1024 x 1024] transposed
    float* __restrict__ f_out,         // [64 x 1024]
    bf16_t* __restrict__ fh_bf) {      // [64 x 1024]
  extern __shared__ char smem[];
  const int lane = threadIdx.x & 31;
  const int wave = threadIdx.x >> 5;
  const int nblk0 = blockIdx.x * 64;

  stage_step_tiles(h_bf, UfT + (size_t)nblk0 * U_DIM, smem, threadIdx.x);
  const bf16_t* sA = (const bf16_t*)smem;                  // [64 x 1024]
  const bf16_t* sB = (const bf16_t*)(smem + LDS_A_BYTES);  // [64 x 1024] (NxK)

  const int m0 = (wave & 3) * 16;
  const int nloc0 = (wave >> 2) * 32;
  v8f acc[2] = {};
  for (int kb0 = 0; kb0 < U_DIM; kb0 += 32) {
    v16bf a = load_a_bf16(sA, U_DIM, m0, kb0, lane);
#pragma unroll
    for (int j = 0; j < 2; ++j) {
      v16bf bfrag = load_b_t(sB, U_DIM, kb0, nloc0 + j * 16, lane);
      acc[j] = __builtin_amdgcn_wmma_f32_16x16x32_bf16(
          false, a, false, bfrag, (short)0, acc[j], false, false);
    }
  }
  const int hlf = lane >> 4;
  const int cidx = lane & 15;
#pragma unroll
  for (int j = 0; j < 2; ++j) {
    const int col = nblk0 + nloc0 + j * 16 + cidx;
#pragma unroll
    for (int v = 0; v < 8; ++v) {
      const int brow = m0 + v + hlf * 8;                       // batch index
      const size_t xidx = ((size_t)brow * T_DIM + t) * U_DIM + col;
      const size_t hidx = (size_t)brow * U_DIM + col;
      const float pre = xf[xidx] + acc[j][v];
      const float f = 1.0f / (1.0f + __expf(-pre));
      f_out[hidx] = f;
      fh_bf[hidx] = f2bf(f * h_f32[hidx]);
    }
  }
}

// Step kernel 2: c = tanh(xc_t + (f*h) @ U_c); h' = h + f*(c - h); write seq out.
__global__ __launch_bounds__(256) void mgu_step_c(
    int t,
    const float* __restrict__ xc,      // [B*T x U]
    const bf16_t* __restrict__ fh_bf,  // [64 x 1024]
    float* __restrict__ h_f32,         // [64 x 1024] (read + write)
    const bf16_t* __restrict__ UcT,    // [1024 x 1024] transposed
    const float* __restrict__ f_in,    // [64 x 1024]
    bf16_t* __restrict__ h_bf,         // [64 x 1024] (write)
    float* __restrict__ out) {         // [B x T x U]
  extern __shared__ char smem[];
  const int lane = threadIdx.x & 31;
  const int wave = threadIdx.x >> 5;
  const int nblk0 = blockIdx.x * 64;

  stage_step_tiles(fh_bf, UcT + (size_t)nblk0 * U_DIM, smem, threadIdx.x);
  const bf16_t* sA = (const bf16_t*)smem;
  const bf16_t* sB = (const bf16_t*)(smem + LDS_A_BYTES);

  const int m0 = (wave & 3) * 16;
  const int nloc0 = (wave >> 2) * 32;
  v8f acc[2] = {};
  for (int kb0 = 0; kb0 < U_DIM; kb0 += 32) {
    v16bf a = load_a_bf16(sA, U_DIM, m0, kb0, lane);
#pragma unroll
    for (int j = 0; j < 2; ++j) {
      v16bf bfrag = load_b_t(sB, U_DIM, kb0, nloc0 + j * 16, lane);
      acc[j] = __builtin_amdgcn_wmma_f32_16x16x32_bf16(
          false, a, false, bfrag, (short)0, acc[j], false, false);
    }
  }
  const int hlf = lane >> 4;
  const int cidx = lane & 15;
#pragma unroll
  for (int j = 0; j < 2; ++j) {
    const int col = nblk0 + nloc0 + j * 16 + cidx;
#pragma unroll
    for (int v = 0; v < 8; ++v) {
      const int brow = m0 + v + hlf * 8;
      const size_t xidx = ((size_t)brow * T_DIM + t) * U_DIM + col;
      const size_t hidx = (size_t)brow * U_DIM + col;
      const float c = tanhf(xc[xidx] + acc[j][v]);
      const float hold = h_f32[hidx];
      const float f = f_in[hidx];
      const float hnew = hold + f * (c - hold);
      h_f32[hidx] = hnew;
      h_bf[hidx] = f2bf(hnew);
      out[xidx] = hnew;
    }
  }
}

// ---------------------------------------------------------------------------
extern "C" void kernel_launch(void* const* d_in, const int* in_sizes, int n_in,
                              void* d_out, int out_size, void* d_ws, size_t ws_size,
                              hipStream_t stream) {
  const float* inputs = (const float*)d_in[0];   // [64,512,512]
  const float* W_f    = (const float*)d_in[1];   // [512,1024]
  const float* U_f    = (const float*)d_in[2];   // [1024,1024]
  const float* b_f    = (const float*)d_in[3];   // [1024]
  const float* W_c    = (const float*)d_in[4];
  const float* U_c    = (const float*)d_in[5];
  const float* b_c    = (const float*)d_in[6];
  float* out = (float*)d_out;                    // [64,512,1024]

  char* ws = (char*)d_ws;
  float*  xf    = (float*)ws;  ws += (size_t)MROWS * U_DIM * sizeof(float);
  float*  xc    = (float*)ws;  ws += (size_t)MROWS * U_DIM * sizeof(float);
  bf16_t* WfT   = (bf16_t*)ws; ws += (size_t)U_DIM * D_DIM * sizeof(bf16_t);
  bf16_t* WcT   = (bf16_t*)ws; ws += (size_t)U_DIM * D_DIM * sizeof(bf16_t);
  bf16_t* UfT   = (bf16_t*)ws; ws += (size_t)U_DIM * U_DIM * sizeof(bf16_t);
  bf16_t* UcT   = (bf16_t*)ws; ws += (size_t)U_DIM * U_DIM * sizeof(bf16_t);
  float*  h_f32 = (float*)ws;  ws += (size_t)B_DIM * U_DIM * sizeof(float);
  bf16_t* h_bf  = (bf16_t*)ws; ws += (size_t)B_DIM * U_DIM * sizeof(bf16_t);
  float*  f_buf = (float*)ws;  ws += (size_t)B_DIM * U_DIM * sizeof(float);
  bf16_t* fh_bf = (bf16_t*)ws; ws += (size_t)B_DIM * U_DIM * sizeof(bf16_t);

  // Allow 256KB dynamic LDS on the step kernels (WGP has 320KB).
  (void)hipFuncSetAttribute((const void*)mgu_step_f,
                            hipFuncAttributeMaxDynamicSharedMemorySize,
                            STEP_LDS_BYTES);
  (void)hipFuncSetAttribute((const void*)mgu_step_c,
                            hipFuncAttributeMaxDynamicSharedMemorySize,
                            STEP_LDS_BYTES);

  // Phase 1: weight conversion + state init.
  mgu_cvt_wt<<<512, 256, 0, stream>>>(W_f, WfT, D_DIM, U_DIM);
  mgu_cvt_wt<<<512, 256, 0, stream>>>(W_c, WcT, D_DIM, U_DIM);
  mgu_cvt_wt<<<1024, 256, 0, stream>>>(U_f, UfT, U_DIM, U_DIM);
  mgu_cvt_wt<<<1024, 256, 0, stream>>>(U_c, UcT, U_DIM, U_DIM);
  mgu_init_h<<<(B_DIM * U_DIM + 255) / 256, 256, 0, stream>>>(h_f32, h_bf);

  // Phase 2: time-parallel input projections (memory-bound WMMA GEMM).
  dim3 pgrid(16, 256, 2);
  mgu_proj<<<pgrid, 256, 0, stream>>>(inputs, WfT, WcT, b_f, b_c, xf, xc);

  // Phase 3: sequential recurrence; stream ordering = device-wide sync.
  for (int t = 0; t < T_DIM; ++t) {
    mgu_step_f<<<16, 256, STEP_LDS_BYTES, stream>>>(t, xf, h_bf, h_f32, UfT,
                                                    f_buf, fh_bf);
    mgu_step_c<<<16, 256, STEP_LDS_BYTES, stream>>>(t, xc, fh_bf, h_f32, UcT,
                                                    f_buf, h_bf, out);
  }
}
